// CIN_54666343743669
// MI455X (gfx1250) — matrix operand order repaired and checked by reading, not compiled, not run on hardware
//
#include <hip/hip_runtime.h>
#include <hip/hip_bf16.h>

// ---------------------------------------------------------------------------
// xDeepFM (CIN + DNN) forward, MI455X / gfx1250, wave32, bf16 WMMA.
//
// Heavy math (CIN contractions + fc1) runs on v_wmma_f32_16x16x32_bf16.
// CIN layer k:  out[bd,o] = sum_{K=k*48+j} (v[bd,k]*u[bd,j]) * W[K,o]
//   -> GEMM with the z row synthesized on the fly into the A fragment.
// All fragment buffers use compile-time-constant indices only (runtime
// ping-pong indices caused scratch spills in a previous revision).
// ---------------------------------------------------------------------------

typedef __attribute__((ext_vector_type(16))) __bf16 v16bf;
typedef __attribute__((ext_vector_type(8)))  float  v8f;

union Frag16 { v16bf v; uint4 q[2]; };   // 32 bytes: one 16-bf16 WMMA operand

#define BN_INVS 0.99950037f   // 1/sqrt(1 + 1e-3)

__device__ __forceinline__ void load_frag(Frag16& f, const __bf16* p) {
    const uint4* q = (const uint4*)p;
    f.q[0] = q[0];
    f.q[1] = q[1];
}

// ---------------------------------------------------------------------------
// Pack a row-major f32 weight [Ktot x ldO] into bf16 WMMA B-fragment order:
//   dst[((c*NT + n)*32 + lane)*16 + e] = W[c*32 + (lane/16)*8 + e + (e>=8 ? 8 : 0), n*16 + lane%16]
// (zero-padded for columns >= O).  Ktot must be a multiple of 32.
// ---------------------------------------------------------------------------
__global__ void pack_b_kernel(const float* __restrict__ w, __bf16* __restrict__ dst,
                              int total, int O, int NT, int ldO) {
    int idx = blockIdx.x * blockDim.x + threadIdx.x;
    if (idx >= total) return;
    int e = idx & 15;
    int l = (idx >> 4) & 31;
    int n = (idx >> 9) % NT;
    int c = idx / (512 * NT);
    int K   = c * 32 + ((l >> 4) << 3) + e + ((e >= 8) ? 8 : 0);
    int col = n * 16 + (l & 15);
    float val = (col < O) ? w[(size_t)K * ldO + col] : 0.0f;
    dst[idx] = (__bf16)val;
}

// ---------------------------------------------------------------------------
// Embedding gather: x0 = emb[x], stored twice in bf16:
//   x0t[(b*128+d)*48 + m]   (CIN layout: one row per (b,d))
//   x0r[b*6144 + m*128 + d] (fc1 layout: reshape(B, M*D))
// ---------------------------------------------------------------------------
__global__ void gather_kernel(const int* __restrict__ x, const float* __restrict__ emb,
                              __bf16* __restrict__ x0t, __bf16* __restrict__ x0r) {
    int tid = blockIdx.x * 256 + threadIdx.x;          // < 512*48*128
    if (tid >= 512 * 48 * 128) return;
    int d = tid & 127;
    int m = (tid >> 7) % 48;
    int b = tid / (48 * 128);
    int row = x[b * 48 + m];
    __bf16 v = (__bf16)emb[row * 128 + d];
    x0t[((size_t)(b * 128 + d)) * 48 + m] = v;
    x0r[(size_t)b * 6144 + m * 128 + d]   = v;
}

// ---------------------------------------------------------------------------
// CIN layer GEMM.  One wave per 16-row bd-tile (4096 tiles).
//   u  : x0t   [65536 x 48]  bf16
//   v  : xk    [65536 x HK]  bf16   (layer 1: v == u)
//   wp : packed weights, K = HK*48, N padded to NT*16
//   xn : output [65536 x O]  bf16
// A fragment (16-bit A 16x32 layout, interleaved by 8):
//   lane half h = lane>>4;  element e -> K = Kbase + h*8 + e + (e>=8 ? 8 : 0)
// B loads are issued before the z-synthesis; the ~100-op LDS/VALU sequence
// hides their latency without needing a second register buffer.
// ---------------------------------------------------------------------------
template <int HK, int O, int NT>
__global__ __launch_bounds__(256) void cin_gemm_kernel(
        const __bf16* __restrict__ u, const __bf16* __restrict__ v,
        const __bf16* __restrict__ wp, __bf16* __restrict__ xn) {
    constexpr int KTOT = HK * 48;          // multiple of 32 for HK in {48,68,32}
    constexpr int KC   = KTOT / 32;
    __shared__ float ldsu[8][16][48];
    __shared__ float ldsv[8][16][HK];

    const int wave = threadIdx.x >> 5;
    const int lane = threadIdx.x & 31;
    const int tile = blockIdx.x * 8 + wave;        // 0..4095
    const int bdBase = tile * 16;
    const int h   = lane >> 4;
    const int row = lane & 15;

    // Stage u, v rows to LDS as f32 (uniform trip counts: no divergence).
    for (int i = lane; i < 16 * 48; i += 32) {
        int r = i / 48, c = i % 48;
        ldsu[wave][r][c] = (float)u[(size_t)(bdBase + r) * 48 + c];
    }
    for (int i = lane; i < 16 * HK; i += 32) {
        int r = i / HK, c = i % HK;
        ldsv[wave][r][c] = (float)v[(size_t)(bdBase + r) * HK + c];
    }
    __syncthreads();

    v8f acc[NT] = {};
    for (int c = 0; c < KC; ++c) {
        // Issue all B-fragment loads first; the z-synthesis below hides them.
        Frag16 bfr[NT];
#pragma unroll
        for (int n = 0; n < NT; ++n)
            load_frag(bfr[n], wp + ((size_t)(c * NT + n) * 32 + lane) * 16);

        // Synthesize this chunk's A fragment from LDS (z = v[k]*u[j]).
        const int Kb = c * 32 + h * 8;
        Frag16 a;
#pragma unroll
        for (int e = 0; e < 16; ++e) {
            int K = Kb + e + ((e >= 8) ? 8 : 0);
            int k = K / 48;
            int j = K - k * 48;
            a.v[e] = (__bf16)(ldsv[wave][row][k] * ldsu[wave][row][j]);
        }

#pragma unroll
        for (int n = 0; n < NT; ++n)
            acc[n] = __builtin_amdgcn_wmma_f32_16x16x32_bf16(
                false, a.v, false, bfr[n].v, (short)0, acc[n], false, false);
    }

    // D layout: VGPR g, lane l -> (row = g + 8*(l>=16), col = l%16)
#pragma unroll
    for (int n = 0; n < NT; ++n) {
        int o = n * 16 + row;
        if (o < O) {
#pragma unroll
            for (int g = 0; g < 8; ++g) {
                int r = g + h * 8;
                xn[(size_t)(bdBase + r) * O + o] = (__bf16)acc[n][g];
            }
        }
    }
}

// ---------------------------------------------------------------------------
// fc1: [512 x 6144] @ [6144 x 64] via WMMA, fused bias + ReLU + BN scale.
// Manual 2x-unrolled ping-pong with *named* fragment sets (compile-time
// indices only) so everything stays in VGPRs: WMMAs on set A overlap the
// loads filling set B, and vice versa.
// ---------------------------------------------------------------------------
__global__ __launch_bounds__(256) void fc1_kernel(
        const __bf16* __restrict__ a, const __bf16* __restrict__ wp,
        const float* __restrict__ bias, const float* __restrict__ gam,
        const float* __restrict__ bet, float* __restrict__ out) {
    const int wave = threadIdx.x >> 5;
    const int lane = threadIdx.x & 31;
    const int tile = blockIdx.x * 8 + wave;        // 0..31
    const int mBase = tile * 16;
    const int h = lane >> 4;
    const int r = lane & 15;

    const __bf16* arow = a + (size_t)(mBase + r) * 6144 + h * 8;
    const __bf16* wlane = wp + (size_t)lane * 16;

    v8f acc[4] = {};
    Frag16 aA, aB;
    Frag16 bA[4], bB[4];

    // Preload chunk 0 into set A.
    aA.q[0] = *(const uint4*)(arow);
    aA.q[1] = *(const uint4*)(arow + 16);
#pragma unroll
    for (int n = 0; n < 4; ++n)
        load_frag(bA[n], wlane + (size_t)n * 512);

    for (int c = 0; c < 192; c += 2) {
        // Prefetch chunk c+1 into set B (c+1 < 192 since 192 is even).
        {
            const __bf16* ap = arow + (c + 1) * 32;
            aB.q[0] = *(const uint4*)(ap);
            aB.q[1] = *(const uint4*)(ap + 16);
#pragma unroll
            for (int n = 0; n < 4; ++n)
                load_frag(bB[n], wlane + ((size_t)(c + 1) * 4 + n) * 512);
        }
        // Consume set A (chunk c).
#pragma unroll
        for (int n = 0; n < 4; ++n)
            acc[n] = __builtin_amdgcn_wmma_f32_16x16x32_bf16(
                false, aA.v, false, bA[n].v, (short)0, acc[n], false, false);

        // Prefetch chunk c+2 into set A (clamped on the last trip).
        {
            const int cp = (c + 2 < 192) ? (c + 2) : 0;
            const __bf16* ap = arow + cp * 32;
            aA.q[0] = *(const uint4*)(ap);
            aA.q[1] = *(const uint4*)(ap + 16);
#pragma unroll
            for (int n = 0; n < 4; ++n)
                load_frag(bA[n], wlane + ((size_t)cp * 4 + n) * 512);
        }
        // Consume set B (chunk c+1).
#pragma unroll
        for (int n = 0; n < 4; ++n)
            acc[n] = __builtin_amdgcn_wmma_f32_16x16x32_bf16(
                false, aB.v, false, bB[n].v, (short)0, acc[n], false, false);
    }

#pragma unroll
    for (int n = 0; n < 4; ++n) {
        int o = n * 16 + r;
#pragma unroll
        for (int g = 0; g < 8; ++g) {
            int rb = mBase + g + h * 8;
            float hv = acc[n][g] + bias[o];
            hv = fmaxf(hv, 0.0f);                               // ReLU
            out[(size_t)rb * 64 + o] = gam[o] * hv * BN_INVS + bet[o];
        }
    }
}

// ---------------------------------------------------------------------------
// Sum-pool CIN layer outputs over D: cin[b, 0:68 | 68:100 | 100:124]
// ---------------------------------------------------------------------------
__global__ void pool_kernel(const __bf16* __restrict__ x1, const __bf16* __restrict__ x2,
                            const __bf16* __restrict__ x3, float* __restrict__ cin) {
    int tid = blockIdx.x * 256 + threadIdx.x;
    if (tid >= 512 * 124) return;
    int b = tid / 124, oo = tid % 124;
    const __bf16* src; int O, o;
    if (oo < 68)       { src = x1; O = 68; o = oo; }
    else if (oo < 100) { src = x2; O = 32; o = oo - 68; }
    else               { src = x3; O = 24; o = oo - 100; }
    float s = 0.0f;
    for (int d = 0; d < 128; ++d)
        s += (float)src[((size_t)(b * 128 + d)) * O + o];
    cin[tid] = s;
}

// fc2: [512x64] @ [64x48], tanh + BN
__global__ void fc2_kernel(const float* __restrict__ d1, const float* __restrict__ w,
                           const float* __restrict__ bias, const float* __restrict__ gam,
                           const float* __restrict__ bet, float* __restrict__ d2) {
    int tid = blockIdx.x * 256 + threadIdx.x;
    if (tid >= 512 * 48) return;
    int b = tid / 48, o = tid % 48;
    float acc = bias[o];
    for (int i = 0; i < 64; ++i) acc += d1[b * 64 + i] * w[i * 48 + o];
    d2[tid] = gam[o] * tanhf(acc) * BN_INVS + bet[o];
}

// fc3: [512x48] @ [48x24], tanh + BN
__global__ void fc3_kernel(const float* __restrict__ d2, const float* __restrict__ w,
                           const float* __restrict__ bias, const float* __restrict__ gam,
                           const float* __restrict__ bet, float* __restrict__ d3) {
    int tid = blockIdx.x * 256 + threadIdx.x;
    if (tid >= 512 * 24) return;
    int b = tid / 24, o = tid % 24;
    float acc = bias[o];
    for (int i = 0; i < 48; ++i) acc += d2[b * 48 + i] * w[i * 24 + o];
    d3[tid] = gam[o] * tanhf(acc) * BN_INVS + bet[o];
}

// Final: concat(deep[24], linear[1], cin[124]) @ out_w + out_b -> sigmoid
__global__ void final_kernel(const int* __restrict__ x, const float* __restrict__ lw,
                             const float* __restrict__ lb, const float* __restrict__ d3,
                             const float* __restrict__ cin, const float* __restrict__ ow,
                             const float* __restrict__ ob, float* __restrict__ out) {
    int b = blockIdx.x * 256 + threadIdx.x;
    if (b >= 512) return;
    float ls = lb[0];
    for (int m = 0; m < 48; ++m) ls += (float)x[b * 48 + m] * lw[m];
    float lin = tanhf(ls);
    float acc = ob[0];
    for (int o = 0; o < 24; ++o) acc += d3[b * 24 + o] * ow[o];
    acc += lin * ow[24];
    for (int i = 0; i < 124; ++i) acc += cin[b * 124 + i] * ow[25 + i];
    out[b] = 1.0f / (1.0f + expf(-acc));
}

// ---------------------------------------------------------------------------
extern "C" void kernel_launch(void* const* d_in, const int* in_sizes, int n_in,
                              void* d_out, int out_size, void* d_ws, size_t ws_size,
                              hipStream_t stream) {
    (void)in_sizes; (void)n_in; (void)out_size; (void)ws_size;

    const int*   x      = (const int*)  d_in[0];
    const float* emb    = (const float*)d_in[1];
    const float* w_cin1 = (const float*)d_in[2];
    const float* w_cin2 = (const float*)d_in[3];
    const float* w_cin3 = (const float*)d_in[4];
    const float* lin_w  = (const float*)d_in[5];
    const float* lin_b  = (const float*)d_in[6];
    const float* fc1_w  = (const float*)d_in[7];
    const float* fc1_b  = (const float*)d_in[8];
    const float* bn1_g  = (const float*)d_in[9];
    const float* bn1_b  = (const float*)d_in[10];
    const float* fc2_w  = (const float*)d_in[11];
    const float* fc2_b  = (const float*)d_in[12];
    const float* bn2_g  = (const float*)d_in[13];
    const float* bn2_b  = (const float*)d_in[14];
    const float* fc3_w  = (const float*)d_in[15];
    const float* fc3_b  = (const float*)d_in[16];
    const float* bn3_g  = (const float*)d_in[17];
    const float* bn3_b  = (const float*)d_in[18];
    const float* out_w  = (const float*)d_in[19];
    const float* out_b  = (const float*)d_in[20];
    float* out = (float*)d_out;

    // Workspace layout (bytes, all 256-aligned). Total ~30.9 MB.
    char* ws = (char*)d_ws;
    __bf16* x0t = (__bf16*)(ws + 0);             // 65536*48*2  = 6291456
    __bf16* x0r = (__bf16*)(ws + 6291456);       // 512*6144*2  = 6291456
    __bf16* xk1 = (__bf16*)(ws + 12582912);      // 65536*68*2  = 8912896
    __bf16* xk2 = (__bf16*)(ws + 21495808);      // 65536*32*2  = 4194304
    __bf16* xk3 = (__bf16*)(ws + 25690112);      // 65536*24*2  = 3145728
    __bf16* wp1 = (__bf16*)(ws + 28835840);      // 72*5*512*2  = 368640
    __bf16* wp2 = (__bf16*)(ws + 29204480);      // 102*2*512*2 = 208896
    __bf16* wp3 = (__bf16*)(ws + 29413376);      // 48*2*512*2  = 98304
    __bf16* wpf = (__bf16*)(ws + 29511680);      // 192*4*512*2 = 786432
    float*  cin = (float*) (ws + 30298112);      // 512*124*4   = 253952
    float*  dp1 = (float*) (ws + 30552064);      // 512*64*4    = 131072
    float*  dp2 = (float*) (ws + 30683136);      // 512*48*4    = 98304
    float*  dp3 = (float*) (ws + 30781440);      // 512*24*4    = 49152

    // 1) Pack weights into WMMA B-fragment order (bf16).
    pack_b_kernel<<<720,  256, 0, stream>>>(w_cin1, wp1, 72 * 5 * 512, 68, 5, 68);
    pack_b_kernel<<<408,  256, 0, stream>>>(w_cin2, wp2, 102 * 2 * 512, 32, 2, 32);
    pack_b_kernel<<<192,  256, 0, stream>>>(w_cin3, wp3, 48 * 2 * 512, 24, 2, 24);
    pack_b_kernel<<<1536, 256, 0, stream>>>(fc1_w,  wpf, 192 * 4 * 512, 64, 4, 64);

    // 2) Embedding gather (both layouts).
    gather_kernel<<<12288, 256, 0, stream>>>(x, emb, x0t, x0r);

    // 3) CIN layers (WMMA).  4096 bd-tiles / 8 waves per block = 512 blocks.
    cin_gemm_kernel<48, 68, 5><<<512, 256, 0, stream>>>(x0t, x0t, wp1, xk1);
    cin_gemm_kernel<68, 32, 2><<<512, 256, 0, stream>>>(x0t, xk1, wp2, xk2);
    cin_gemm_kernel<32, 24, 2><<<512, 256, 0, stream>>>(x0t, xk2, wp3, xk3);

    // 4) Sum-pool over D -> cin[512,124]
    pool_kernel<<<248, 256, 0, stream>>>(xk1, xk2, xk3, cin);

    // 5) Deep path: fc1 (WMMA) -> fc2 -> fc3
    fc1_kernel<<<4,  256, 0, stream>>>(x0r, wpf, fc1_b, bn1_g, bn1_b, dp1);
    fc2_kernel<<<96, 256, 0, stream>>>(dp1, fc2_w, fc2_b, bn2_g, bn2_b, dp2);
    fc3_kernel<<<48, 256, 0, stream>>>(dp2, fc3_w, fc3_b, bn3_g, bn3_b, dp3);

    // 6) Linear + concat + sigmoid
    final_kernel<<<2, 256, 0, stream>>>(x, lin_w, lin_b, dp3, cin, out_w, out_b, out);
}
